// MultiHeadAttention_88330297409721
// MI455X (gfx1250) — matrix operand order repaired and checked by reading, not compiled, not run on hardware
//
#include <hip/hip_runtime.h>

// MHA: B=2, S=2048, D=1024, H=16, HD=64.
// Tensor-core bound -> bf16 WMMA (v_wmma_f32_16x16x32_bf16), f32 accumulate.
// Async global->LDS staging (ASYNCcnt path) + double-buffered tiles.
#define Dm   1024
#define Hn   16
#define HDn  64
#define Bn   2
#define Sn   2048
#define Mrows (Bn * Sn)   // 4096

typedef __attribute__((ext_vector_type(16))) __bf16 v16bf;
typedef __attribute__((ext_vector_type(8)))  float  v8f;
typedef __attribute__((ext_vector_type(4)))  int    v4i_;

union FragBF { v16bf v; uint4 q[2]; };

__device__ __forceinline__ unsigned short f2bf(float f) {
  unsigned int u = __float_as_uint(f);
  u += 0x7FFFu + ((u >> 16) & 1u);           // round-to-nearest-even
  return (unsigned short)(u >> 16);
}
__device__ __forceinline__ unsigned int pack2bf(float lo, float hi) {
  return (unsigned int)f2bf(lo) | ((unsigned int)f2bf(hi) << 16);
}

// ---- CDNA5 async global->LDS (direct LDS write, tracked by ASYNCcnt) -------
#if defined(__AMDGCN__) && __has_builtin(__builtin_amdgcn_global_load_async_to_lds_b128)
#define HAVE_ASYNC_LDS 1
#else
#define HAVE_ASYNC_LDS 0
#endif

__device__ __forceinline__ void async_copy_b128(uint4* lds_dst, const uint4* gsrc) {
#if HAVE_ASYNC_LDS
  v4i_* g_gen = (v4i_*)const_cast<uint4*>(gsrc);   // pointee change, generic AS
  v4i_* l_gen = (v4i_*)lds_dst;
  __builtin_amdgcn_global_load_async_to_lds_b128(
      (__attribute__((address_space(1))) v4i_*)g_gen,   // flat -> global
      (__attribute__((address_space(3))) v4i_*)l_gen,   // flat -> LDS offset
      0, 0);
#else
  *lds_dst = *gsrc;
#endif
}
__device__ __forceinline__ void async_wait_all() {
#if HAVE_ASYNC_LDS
#if __has_builtin(__builtin_amdgcn_s_wait_asynccnt)
  __builtin_amdgcn_s_wait_asynccnt(0);
#else
  asm volatile("s_wait_asynccnt 0x0" ::: "memory");
#endif
#endif
}

// ---------------------------------------------------------------------------
// Prep: WT_bf16[n][k] = bf16(W[k][n])   (GEMM B tiles then stream contiguously)
// ---------------------------------------------------------------------------
__global__ __launch_bounds__(256) void transpose_convert_w(
    const float* __restrict__ W, unsigned short* __restrict__ WT) {
  __shared__ float tile[32][33];
  int n0 = blockIdx.x * 32, k0 = blockIdx.y * 32;
  int tx = threadIdx.x, ty = threadIdx.y;          // 32 x 8
  for (int i = 0; i < 4; ++i) {
    int k = ty + i * 8;
    tile[k][tx] = W[(size_t)(k0 + k) * Dm + (n0 + tx)];
  }
  __syncthreads();
  for (int i = 0; i < 4; ++i) {
    int n = ty + i * 8;
    WT[(size_t)(n0 + n) * Dm + (k0 + tx)] = f2bf(tile[tx][n]);
  }
}

// ---------------------------------------------------------------------------
// Tiled GEMM, BM=BN=128, BK=32, double-buffered LDS, 8 waves (2x4 wave grid),
// each wave computes 4x2 WMMA subtiles (64x32 of C).
// MODE 0: A = x (f32, convert on stage), out bf16 per-head [b][h][s][hd] + bias
// MODE 1: A = ctx (bf16, async stage),  out f32 [m][n] + bias
// LDS rows padded to 5 uint4 (80B) -> conflict-free b128 fragment reads.
// ---------------------------------------------------------------------------
template <int MODE>
__global__ __launch_bounds__(256) void gemm128(
    const void* __restrict__ Aptr, const unsigned short* __restrict__ WT,
    const float* __restrict__ bias, void* __restrict__ Out) {
  __shared__ uint4 AsU4[2][128 * 5];
  __shared__ uint4 BsU4[2][128 * 5];
  const int tid  = threadIdx.x;
  const int lane = tid & 31, wave = tid >> 5;
  const int half = lane >> 4, l15 = lane & 15;
  const int wave_m = wave >> 2, wave_n = wave & 3;
  const int m0 = blockIdx.y * 128, n0 = blockIdx.x * 128;

  auto stageA = [&](int bufi, int k0) {
    if (MODE == 0) {  // f32 -> bf16 conversion while staging (VALU path)
      const float* A = (const float*)Aptr;
      for (int i = 0; i < 4; ++i) {
        int idx = i * 256 + tid;               // 1024 float4 per tile
        int row = idx >> 3, c4 = idx & 7;
        const float4 f = ((const float4*)(A + (size_t)(m0 + row) * Dm + k0))[c4];
        uint2 p; p.x = pack2bf(f.x, f.y); p.y = pack2bf(f.z, f.w);
        ((uint2*)AsU4[bufi])[row * 10 + c4] = p; // row stride 5 uint4 = 10 uint2
      }
    } else {          // bf16: async direct-to-LDS
      const uint4* A = (const uint4*)Aptr;     // 128 uint4 per row
      for (int i = 0; i < 2; ++i) {
        int idx = i * 256 + tid;               // 512 uint4 per tile
        int row = idx >> 2, c = idx & 3;
        async_copy_b128(&AsU4[bufi][row * 5 + c],
                        &A[(size_t)(m0 + row) * 128 + (k0 >> 3) + c]);
      }
    }
  };
  auto stageB = [&](int bufi, int k0) {
    const uint4* Wt4 = (const uint4*)WT;
    for (int i = 0; i < 2; ++i) {
      int idx = i * 256 + tid;
      int row = idx >> 2, c = idx & 3;
      async_copy_b128(&BsU4[bufi][row * 5 + c],
                      &Wt4[(size_t)(n0 + row) * 128 + (k0 >> 3) + c]);
    }
  };

  v8f acc[4][2];
  for (int mi = 0; mi < 4; ++mi)
    for (int ni = 0; ni < 2; ++ni)
      for (int r = 0; r < 8; ++r) acc[mi][ni][r] = 0.f;

  stageA(0, 0);
  stageB(0, 0);
  async_wait_all();
  __syncthreads();

  int buf = 0;
  for (int k0 = 0; k0 < Dm; k0 += 32) {
    const int kn = k0 + 32;
    if (kn < Dm) { stageA(buf ^ 1, kn); stageB(buf ^ 1, kn); }  // prefetch

    FragBF a[4], b[2];
    for (int mi = 0; mi < 4; ++mi) {
      int m = wave_m * 64 + mi * 16 + l15;
      a[mi].q[0] = AsU4[buf][m * 5 + half];         // K 0-7 | 8-15
      a[mi].q[1] = AsU4[buf][m * 5 + 2 + half];     // K 16-23 | 24-31
    }
    for (int ni = 0; ni < 2; ++ni) {
      int n = wave_n * 32 + ni * 16 + l15;
      b[ni].q[0] = BsU4[buf][n * 5 + half];         // K 0-7 | 8-15
      b[ni].q[1] = BsU4[buf][n * 5 + 2 + half];     // K 16-23 | 24-31
    }
    for (int mi = 0; mi < 4; ++mi)
      for (int ni = 0; ni < 2; ++ni)
        acc[mi][ni] = __builtin_amdgcn_wmma_f32_16x16x32_bf16(
            false, a[mi].v, false, b[ni].v, (short)0, acc[mi][ni], false, false);

    async_wait_all();        // next tile landed in LDS
    __syncthreads();
    buf ^= 1;
  }

  float bv[2];
  for (int ni = 0; ni < 2; ++ni)
    bv[ni] = bias[n0 + wave_n * 32 + ni * 16 + l15];
  for (int mi = 0; mi < 4; ++mi)
    for (int ni = 0; ni < 2; ++ni) {
      int n = n0 + wave_n * 32 + ni * 16 + l15;
      for (int r = 0; r < 8; ++r) {
        int m = m0 + wave_m * 64 + mi * 16 + half * 8 + r;
        float v = acc[mi][ni][r] + bv[ni];
        if (MODE == 0) {
          int bb = m >> 11, s = m & 2047, h = n >> 6, hd = n & 63;
          ((unsigned short*)Out)[(((size_t)bb * Hn + h) * Sn + s) * HDn + hd] = f2bf(v);
        } else {
          ((float*)Out)[(size_t)m * Dm + n] = v;
        }
      }
    }
}

// ---------------------------------------------------------------------------
// Flash attention: block = 128 queries of one (b,h); 8 waves x 16 queries.
// Each wave owns full 128-key rows -> online softmax stays inside the wave
// (16-lane xor shuffles within wave32 halves, per the C-matrix row layout).
// K/V tiles double-buffered; K staged async, V transposed through VALU.
// ---------------------------------------------------------------------------
__global__ __launch_bounds__(256) void flash_attn(
    const unsigned short* __restrict__ Qg, const unsigned short* __restrict__ Kg,
    const unsigned short* __restrict__ Vg, unsigned short* __restrict__ ctx) {
  __shared__ uint4 Qs[128 * 9];       // [q][hd]   rows padded 8->9 uint4
  __shared__ uint4 Ks[2][128 * 9];    // [key][hd]
  __shared__ uint4 Vt[2][64 * 17];    // [hd][key] rows padded 16->17 uint4
  __shared__ uint4 Pl[128 * 17];      // [q][key]  bf16 probabilities

  const int tid  = threadIdx.x;
  const int lane = tid & 31, wave = tid >> 5;
  const int half = lane >> 4, l15 = lane & 15;
  const int q0 = blockIdx.x * 128;
  const int bh = blockIdx.y;                       // b*H + h
  const size_t head_base = (size_t)bh * Sn * HDn;
  const int qrow = wave * 16;

  auto stageKV = [&](int bufi, int kv0) {
    {  // K tile: async direct-to-LDS (contiguous 16KB, padded rows)
      const uint4* src = (const uint4*)(Kg + head_base + (size_t)kv0 * HDn);
      for (int i = 0; i < 4; ++i) {
        int idx = i * 256 + tid;
        async_copy_b128(&Ks[bufi][(idx >> 3) * 9 + (idx & 7)], &src[idx]);
      }
    }
    {  // V tile transposed: Vt[hd][key] (needs VALU shuffle -> sync path)
      const unsigned int* src = (const unsigned int*)(Vg + head_base + (size_t)kv0 * HDn);
      unsigned short* vt = (unsigned short*)Vt[bufi];
      for (int i = 0; i < 16; ++i) {
        int idx = i * 256 + tid;                   // coalesced along hd
        int key = idx >> 5, hp = idx & 31;
        unsigned int d = src[key * 32 + hp];
        vt[(hp * 2) * 136 + key]     = (unsigned short)(d & 0xFFFF);
        vt[(hp * 2 + 1) * 136 + key] = (unsigned short)(d >> 16);
      }
    }
  };

  {  // stage Q (async) + first K/V tile
    const uint4* qsrc = (const uint4*)(Qg + head_base + (size_t)q0 * HDn);
    for (int i = 0; i < 4; ++i) {
      int idx = i * 256 + tid;
      async_copy_b128(&Qs[(idx >> 3) * 9 + (idx & 7)], &qsrc[idx]);
    }
  }
  stageKV(0, 0);
  async_wait_all();
  __syncthreads();

  float m_run[8], l_run[8];
  for (int r = 0; r < 8; ++r) { m_run[r] = -1e30f; l_run[r] = 0.f; }
  v8f o_acc[4];
  for (int gn = 0; gn < 4; ++gn)
    for (int r = 0; r < 8; ++r) o_acc[gn][r] = 0.f;
  const float scale = 0.125f;                      // 1/sqrt(64)

  int buf = 0;
  for (int kv0 = 0; kv0 < Sn; kv0 += 128) {
    if (kv0 + 128 < Sn) stageKV(buf ^ 1, kv0 + 128);   // prefetch next tile

    // ---- S = Q K^T * scale : 16 queries x 128 keys (2 ksteps x 8 groups)
    v8f s_acc[8];
    for (int g = 0; g < 8; ++g)
      for (int r = 0; r < 8; ++r) s_acc[g][r] = 0.f;
    {
      const int q = qrow + l15;
      for (int ks = 0; ks < 2; ++ks) {
        FragBF aq, bk[8];
        aq.q[0] = Qs[q * 9 + ks * 4 + half];
        aq.q[1] = Qs[q * 9 + ks * 4 + 2 + half];
        for (int g = 0; g < 8; ++g) {            // preload all B frags
          int n = g * 16 + l15;
          bk[g].q[0] = Ks[buf][n * 9 + ks * 4 + half];
          bk[g].q[1] = Ks[buf][n * 9 + ks * 4 + 2 + half];
        }
        for (int g = 0; g < 8; ++g)
          s_acc[g] = __builtin_amdgcn_wmma_f32_16x16x32_bf16(
              false, aq.v, false, bk[g].v, (short)0, s_acc[g], false, false);
      }
    }
    for (int g = 0; g < 8; ++g)
      for (int r = 0; r < 8; ++r) s_acc[g][r] *= scale;

    // ---- online softmax (row = qrow + half*8 + r lives in one 16-lane half)
    float corr[8];
    for (int r = 0; r < 8; ++r) {
      float mx = s_acc[0][r];
      for (int g = 1; g < 8; ++g) mx = fmaxf(mx, s_acc[g][r]);
      for (int off = 8; off >= 1; off >>= 1) mx = fmaxf(mx, __shfl_xor(mx, off, 32));
      float m_new = fmaxf(m_run[r], mx);
      corr[r] = __expf(m_run[r] - m_new);
      float rs = 0.f;
      for (int g = 0; g < 8; ++g) {
        float p = __expf(s_acc[g][r] - m_new);
        s_acc[g][r] = p;
        rs += p;
      }
      for (int off = 8; off >= 1; off >>= 1) rs += __shfl_xor(rs, off, 32);
      l_run[r] = l_run[r] * corr[r] + rs;
      m_run[r] = m_new;
    }
    for (int gn = 0; gn < 4; ++gn)
      for (int r = 0; r < 8; ++r) o_acc[gn][r] *= corr[r];

    // ---- P -> LDS (own rows only; per-wave LDS ordering, no barrier needed)
    {
      unsigned short* pw = (unsigned short*)Pl;
      for (int g = 0; g < 8; ++g)
        for (int r = 0; r < 8; ++r) {
          int q = qrow + half * 8 + r;
          pw[q * 136 + g * 16 + l15] = f2bf(s_acc[g][r]);
        }
    }

    // ---- O += P V : K-dim = 128 keys (4 ksteps), N = 64 hd (4 groups)
    for (int ks = 0; ks < 4; ++ks) {
      FragBF ap, bvv[4];
      const int q = qrow + l15;
      ap.q[0] = Pl[q * 17 + ks * 4 + half];
      ap.q[1] = Pl[q * 17 + ks * 4 + 2 + half];
      for (int gn = 0; gn < 4; ++gn) {           // preload all B frags
        int n = gn * 16 + l15;
        bvv[gn].q[0] = Vt[buf][n * 17 + ks * 4 + half];
        bvv[gn].q[1] = Vt[buf][n * 17 + ks * 4 + 2 + half];
      }
      for (int gn = 0; gn < 4; ++gn)
        o_acc[gn] = __builtin_amdgcn_wmma_f32_16x16x32_bf16(
            false, ap.v, false, bvv[gn].v, (short)0, o_acc[gn], false, false);
    }

    async_wait_all();         // prefetched K tile landed
    __syncthreads();
    buf ^= 1;
  }

  // normalize, write ctx bf16 [b][s][h*64+hd]
  const int b = bh >> 4, h = bh & 15;
  for (int gn = 0; gn < 4; ++gn)
    for (int r = 0; r < 8; ++r) {
      int q = q0 + qrow + half * 8 + r;
      float v = o_acc[gn][r] / l_run[r];
      ctx[((size_t)b * Sn + q) * Dm + h * HDn + gn * 16 + l15] = f2bf(v);
    }
}

// ---------------------------------------------------------------------------
extern "C" void kernel_launch(void* const* d_in, const int* in_sizes, int n_in,
                              void* d_out, int out_size, void* d_ws, size_t ws_size,
                              hipStream_t stream) {
  (void)in_sizes; (void)n_in; (void)out_size; (void)ws_size;
  const float* x  = (const float*)d_in[0];
  const float* Wq = (const float*)d_in[1];
  const float* bq = (const float*)d_in[2];
  const float* Wk = (const float*)d_in[3];
  const float* bk = (const float*)d_in[4];
  const float* Wv = (const float*)d_in[5];
  const float* bv = (const float*)d_in[6];
  const float* Wo = (const float*)d_in[7];
  const float* bo = (const float*)d_in[8];
  float* out = (float*)d_out;

  char* ws = (char*)d_ws;
  const size_t szBig = (size_t)Mrows * Dm * sizeof(unsigned short);   // 8 MB
  const size_t szW   = (size_t)Dm * Dm * sizeof(unsigned short);      // 2 MB
  unsigned short* Qb  = (unsigned short*)(ws + 0 * szBig);
  unsigned short* Kb  = (unsigned short*)(ws + 1 * szBig);
  unsigned short* Vb  = (unsigned short*)(ws + 2 * szBig);
  unsigned short* Cb  = (unsigned short*)(ws + 3 * szBig);
  unsigned short* WTq = (unsigned short*)(ws + 4 * szBig);
  unsigned short* WTk = (unsigned short*)((char*)WTq + szW);
  unsigned short* WTv = (unsigned short*)((char*)WTk + szW);
  unsigned short* WTo = (unsigned short*)((char*)WTv + szW);

  dim3 tb(32, 8), tg(Dm / 32, Dm / 32);
  transpose_convert_w<<<tg, tb, 0, stream>>>(Wq, WTq);
  transpose_convert_w<<<tg, tb, 0, stream>>>(Wk, WTk);
  transpose_convert_w<<<tg, tb, 0, stream>>>(Wv, WTv);
  transpose_convert_w<<<tg, tb, 0, stream>>>(Wo, WTo);

  dim3 gg(Dm / 128, Mrows / 128);   // (8, 32)
  gemm128<0><<<gg, 256, 0, stream>>>(x, WTq, bq, Qb);
  gemm128<0><<<gg, 256, 0, stream>>>(x, WTk, bk, Kb);
  gemm128<0><<<gg, 256, 0, stream>>>(x, WTv, bv, Vb);

  flash_attn<<<dim3(Sn / 128, Bn * Hn), 256, 0, stream>>>(Qb, Kb, Vb, Cb);

  gemm128<1><<<gg, 256, 0, stream>>>(Cb, WTo, bo, out);
}
